// GraphPooling_61340722921904
// MI455X (gfx1250) — matrix operand order repaired and checked by reading, not compiled, not run on hardware
//
#include <hip/hip_runtime.h>
#include <hip/hip_bf16.h>
#include <math.h>
#include <stdint.h>

typedef __bf16 bf16;
typedef __attribute__((ext_vector_type(16))) __bf16 v16bf;
typedef __attribute__((ext_vector_type(8)))  float  v8f;

#define BB   8
#define SLN  1024
#define CCH  768
#define KCT  64
#define NHD  12
#define HDD  64
#define NPD  1025          // SL+1
#define ATT_SCALE 0.125f   // HD^-0.5

// ---------------------------------------------------------------------------
// helpers
// ---------------------------------------------------------------------------
__device__ inline v8f vzero8() {
    v8f v;
#pragma unroll
    for (int i = 0; i < 8; ++i) v[i] = 0.0f;
    return v;
}

__device__ inline v8f wmma_bf16(v16bf a, v16bf b, v8f c) {
    // D = A(16x32) x B(32x16) + C, f32 accumulate
    return __builtin_amdgcn_wmma_f32_16x16x32_bf16(false, a, false, b, (short)0, c, false, false);
}

// async copy 16B global -> LDS (per lane), tracked by ASYNCcnt
__device__ inline void async_copy16(unsigned lds_off, const void* gptr) {
    unsigned long long ga = (unsigned long long)(uintptr_t)gptr;
    asm volatile("global_load_async_to_lds_b128 %0, %1, off"
                 :: "v"(lds_off), "v"(ga) : "memory");
}
__device__ inline void wait_async0() {
    asm volatile("s_wait_asynccnt 0x0" ::: "memory");
}

// A-fragment (16x32 bf16) from row-major [rows][ld] at (row0, k0).
// lanes 0-15: elems 0-7 -> K 0..7,  elems 8-15 -> K 16..23
// lanes16-31: elems 0-7 -> K 8..15, elems 8-15 -> K 24..31
__device__ inline v16bf lds_afrag(const bf16* base, int row0, int ld, int k0, int lm, int lh) {
    v16bf f;
    const bf16* p = base + (row0 + lm) * ld + k0 + lh * 8;
    *(uint4*)&f       = *(const uint4*)p;
    *((uint4*)&f + 1) = *(const uint4*)(p + 16);
    return f;
}

// B-fragment (32x16 bf16) from TRANSPOSED storage Bt[n][k] row-major ld=ldt.
// lanes 0-15: elems 0-7 -> K 0..7, 8-15 -> K 16..23 ; lanes 16-31: K 8..15, K 24..31 ; n = lane%16
__device__ inline v16bf lds_bfrag(const bf16* baseT, int n0, int ldt, int k0, int lm, int lh) {
    v16bf f;
    const bf16* p = baseT + (n0 + lm) * ldt + k0 + lh * 8;
    *(uint4*)&f       = *(const uint4*)p;
    *((uint4*)&f + 1) = *(const uint4*)(p + 16);
    return f;
}

__device__ inline float rowred_max16(float v) {
    v = fmaxf(v, __shfl_xor(v, 1, 32));
    v = fmaxf(v, __shfl_xor(v, 2, 32));
    v = fmaxf(v, __shfl_xor(v, 4, 32));
    v = fmaxf(v, __shfl_xor(v, 8, 32));
    return v;
}
__device__ inline float rowred_sum16(float v) {
    v += __shfl_xor(v, 1, 32);
    v += __shfl_xor(v, 2, 32);
    v += __shfl_xor(v, 4, 32);
    v += __shfl_xor(v, 8, 32);
    return v;
}

__device__ inline float block_reduce_sum(float v, float* sb) {
    int tid = threadIdx.x;
    sb[tid] = v;
    __syncthreads();
    for (int s = 128; s > 0; s >>= 1) {
        if (tid < s) sb[tid] += sb[tid + s];
        __syncthreads();
    }
    float r = sb[0];
    __syncthreads();
    return r;
}

// ---------------------------------------------------------------------------
// small utility kernels
// ---------------------------------------------------------------------------
__global__ void f32_to_bf16_kernel(const float* __restrict__ in, bf16* __restrict__ out, int n) {
    int i = blockIdx.x * 256 + threadIdx.x;
    if (i < n) out[i] = (bf16)in[i];
}

__global__ void zero_f32_kernel(float* p, int n) {
    int i = blockIdx.x * 256 + threadIdx.x;
    if (i < n) p[i] = 0.0f;
}

// column mean over rows: in [B][R][C] -> out [B][C]; optional bool mask [B][R]
__global__ void colmean_kernel(const float* __restrict__ in, const unsigned char* __restrict__ mask,
                               float* __restrict__ out, int R, int C) {
    int b = blockIdx.y;
    int c = blockIdx.x * 256 + threadIdx.x;
    if (c >= C) return;
    const float* p = in + (size_t)b * R * C + c;
    float s = 0.0f;
    if (mask) {
        const unsigned char* mp = mask + (size_t)b * R;
        int valid = 0;
        for (int r = 0; r < R; ++r) {
            float v = p[(size_t)r * C];
            if (!mp[r]) { s += v; valid++; }
        }
        if (valid < 1) valid = 1;
        out[(size_t)b * C + c] = s / (float)valid;
    } else {
        for (int r = 0; r < R; ++r) s += p[(size_t)r * C];
        out[(size_t)b * C + c] = s / (float)R;
    }
}

// sampling_src = l2norm(concat(mean, filled_src)) stored TRANSPOSED: [B][C][NP]
__global__ __launch_bounds__(256) void build_samplingT_kernel(const float* __restrict__ src,
                                                              const unsigned char* __restrict__ mask,
                                                              const float* __restrict__ mean,
                                                              float* __restrict__ samplingT) {
    int rowg = blockIdx.x;
    int b = rowg / NPD, j = rowg % NPD;
    __shared__ float buf[CCH];
    __shared__ float red[256];
    int tid = threadIdx.x;
    float ss = 0.0f;
    for (int c = tid; c < CCH; c += 256) {
        float v;
        if (j == 0) v = mean[b * CCH + c];
        else {
            unsigned char mk = mask[(size_t)b * SLN + (j - 1)];
            v = mk ? mean[b * CCH + c] : src[((size_t)b * SLN + (j - 1)) * CCH + c];
        }
        buf[c] = v;
        ss += v * v;
    }
    ss = block_reduce_sum(ss, red);
    float inv = 1.0f / fmaxf(sqrtf(ss), 1e-12f);
    for (int c = tid; c < CCH; c += 256)
        samplingT[(size_t)b * CCH * NPD + (size_t)c * NPD + j] = buf[c] * inv;
}

// farthest point sampling (one block per batch)
__global__ __launch_bounds__(1024) void fps_kernel(const float* __restrict__ samplingT,
                                                   int* __restrict__ inds_i,
                                                   float* __restrict__ out_inds) {
    int b = blockIdx.x, tid = threadIdx.x;
    const float* P = samplingT + (size_t)b * CCH * NPD;
    __shared__ float dist[NPD];
    __shared__ float lastPt[CCH];
    __shared__ float red[1024];
    __shared__ int   redi[1024];
    __shared__ int   indsh[KCT + 1];
    __shared__ int   lastIdx;
    for (int i = tid; i < NPD; i += 1024) dist[i] = 1e30f;
    if (tid == 0) { lastIdx = 0; indsh[0] = 0; }
    __syncthreads();
    for (int t = 1; t <= KCT; ++t) {
        int li = lastIdx;
        for (int c = tid; c < CCH; c += 1024) lastPt[c] = P[(size_t)c * NPD + li];
        __syncthreads();
        float best = -1.0f; int bi = 0;
        for (int i = tid; i < NPD; i += 1024) {
            float s = 0.0f;
            for (int c = 0; c < CCH; ++c) {
                float d = P[(size_t)c * NPD + i] - lastPt[c];
                s += d * d;
            }
            float dm = fminf(dist[i], s);
            dist[i] = dm;
            if (dm > best) { best = dm; bi = i; }
        }
        red[tid] = best; redi[tid] = bi;
        __syncthreads();
        for (int s = 512; s > 0; s >>= 1) {
            if (tid < s) {
                if (red[tid + s] > red[tid] ||
                    (red[tid + s] == red[tid] && redi[tid + s] < redi[tid])) {
                    red[tid] = red[tid + s]; redi[tid] = redi[tid + s];
                }
            }
            __syncthreads();
        }
        if (tid == 0) { lastIdx = redi[0]; indsh[t] = redi[0]; }
        __syncthreads();
    }
    if (tid < KCT) {
        int v = indsh[tid + 1] - 1;
        inds_i[b * KCT + tid] = v;
        out_inds[b * KCT + tid] = (float)v;
    }
}

// LayerNorm rows of f32 [rows][C] -> bf16
__global__ __launch_bounds__(256) void ln_rows_kernel(const float* __restrict__ in,
                                                      const float* __restrict__ g,
                                                      const float* __restrict__ be,
                                                      bf16* __restrict__ out, int C, float eps) {
    int row = blockIdx.x, tid = threadIdx.x;
    __shared__ float buf[3072];
    __shared__ float red[256];
    const float* p = in + (size_t)row * C;
    float s = 0.0f, sq = 0.0f;
    for (int c = tid; c < C; c += 256) { float v = p[c]; buf[c] = v; s += v; sq += v * v; }
    s  = block_reduce_sum(s, red);
    sq = block_reduce_sum(sq, red);
    float mean = s / C;
    float var  = sq / C - mean * mean;
    float inv  = rsqrtf(var + eps);
    for (int c = tid; c < C; c += 256)
        out[(size_t)row * C + c] = (bf16)((buf[c] - mean) * inv * g[c] + be[c]);
}

// LayerNorm of concat(cls, src) rows -> bf16 [B*(SL+1)][C]
__global__ __launch_bounds__(256) void ln_cat_kernel(const float* __restrict__ cls,
                                                     const float* __restrict__ src,
                                                     const float* __restrict__ g,
                                                     const float* __restrict__ be,
                                                     bf16* __restrict__ out, float eps) {
    int row = blockIdx.x, tid = threadIdx.x;
    int b = row / NPD, j = row % NPD;
    __shared__ float buf[CCH];
    __shared__ float red[256];
    const float* p = (j == 0) ? (cls + (size_t)b * CCH)
                              : (src + ((size_t)b * SLN + (j - 1)) * CCH);
    float s = 0.0f, sq = 0.0f;
    for (int c = tid; c < CCH; c += 256) { float v = p[c]; buf[c] = v; s += v; sq += v * v; }
    s  = block_reduce_sum(s, red);
    sq = block_reduce_sum(sq, red);
    float mean = s / CCH, var = sq / CCH - mean * mean;
    float inv = rsqrtf(var + eps);
    for (int c = tid; c < CCH; c += 256)
        out[(size_t)row * CCH + c] = (bf16)((buf[c] - mean) * inv * g[c] + be[c]);
}

// LayerNorm of concat(f1_cls(bf16), centroids(f32)) -> bf16 [B*65][3072]
__global__ __launch_bounds__(256) void ln_cat2_kernel(const bf16* __restrict__ f1,
                                                      const float* __restrict__ cent,
                                                      const float* __restrict__ g,
                                                      const float* __restrict__ be,
                                                      bf16* __restrict__ out, float eps) {
    int row = blockIdx.x, tid = threadIdx.x;
    int b = row / (KCT + 1), j = row % (KCT + 1);
    const int C = 4 * CCH;
    __shared__ float buf[3072];
    __shared__ float red[256];
    float s = 0.0f, sq = 0.0f;
    for (int c = tid; c < C; c += 256) {
        float v = (j == 0) ? (float)f1[((size_t)b * NPD) * C + c]
                           : cent[((size_t)b * KCT + (j - 1)) * C + c];
        buf[c] = v; s += v; sq += v * v;
    }
    s  = block_reduce_sum(s, red);
    sq = block_reduce_sum(sq, red);
    float mean = s / C, var = sq / C - mean * mean;
    float inv = rsqrtf(var + eps);
    for (int c = tid; c < C; c += 256)
        out[(size_t)row * C + c] = (bf16)((buf[c] - mean) * inv * g[c] + be[c]);
}

// node_features = x - colmean(x) + bias ; write l2-normalized bf16
__global__ __launch_bounds__(256) void node_norm_kernel(const float* __restrict__ x,
                                                        const float* __restrict__ xmean,
                                                        const float* __restrict__ bbias,
                                                        bf16* __restrict__ out) {
    int row = blockIdx.x, tid = threadIdx.x;
    int b = row >> 10;
    __shared__ float buf[CCH];
    __shared__ float red[256];
    float ss = 0.0f;
    for (int c = tid; c < CCH; c += 256) {
        float v = x[(size_t)row * CCH + c] - xmean[b * CCH + c] + bbias[c];
        buf[c] = v; ss += v * v;
    }
    ss = block_reduce_sum(ss, red);
    float inv = 1.0f / fmaxf(sqrtf(ss), 1e-12f);
    for (int c = tid; c < CCH; c += 256)
        out[(size_t)row * CCH + c] = (bf16)(buf[c] * inv);
}

// gather l2-normalized centroid features, transposed: centT[b][c][k]
__global__ __launch_bounds__(256) void gather_centT_kernel(const bf16* __restrict__ node_bf,
                                                           const int* __restrict__ inds_i,
                                                           bf16* __restrict__ centT) {
    int b = blockIdx.x >> 6, k = blockIdx.x & 63;
    int ind = inds_i[b * KCT + k];
    for (int c = threadIdx.x; c < CCH; c += 256)
        centT[(size_t)b * CCH * KCT + (size_t)c * KCT + k] =
            node_bf[((size_t)b * SLN + ind) * CCH + c];
}

// softmax over K=64 per (b,n); write assignT bf16 [B][K][SL] and colsum [B][K]
__global__ __launch_bounds__(256) void softmax_assign_kernel(const float* __restrict__ logits,
                                                             bf16* __restrict__ assignT,
                                                             float* __restrict__ colsum) {
    int tid = threadIdx.x, wave = tid >> 5, lane = tid & 31;
    int row = blockIdx.x * 8 + wave;
    int b = row >> 10, n = row & 1023;
    const float* lp = logits + (size_t)row * KCT;
    float l1 = lp[lane], l2 = lp[lane + 32];
    float m = fmaxf(l1, l2);
    for (int d = 1; d < 32; d <<= 1) m = fmaxf(m, __shfl_xor(m, d, 32));
    float e1 = __expf(l1 - m), e2 = __expf(l2 - m);
    float s = e1 + e2;
    for (int d = 1; d < 32; d <<= 1) s += __shfl_xor(s, d, 32);
    float inv = 1.0f / s;
    float p1 = e1 * inv, p2 = e2 * inv;
    assignT[((size_t)b * KCT + lane) * SLN + n]        = (bf16)p1;
    assignT[((size_t)b * KCT + lane + 32) * SLN + n]   = (bf16)p2;
    atomicAdd(&colsum[b * KCT + lane], p1);
    atomicAdd(&colsum[b * KCT + lane + 32], p2);
}

// final: cls_out, centroids_out (+gathered src)
__global__ __launch_bounds__(256) void final_out_kernel(const float* __restrict__ f2,
                                                        const float* __restrict__ cls,
                                                        const float* __restrict__ src,
                                                        const int* __restrict__ inds_i,
                                                        float* __restrict__ out_cls,
                                                        float* __restrict__ out_cent) {
    int blk = blockIdx.x;
    int b = blk / (KCT + 1), j = blk % (KCT + 1);
    if (j == 0) {
        for (int c = threadIdx.x; c < CCH; c += 256)
            out_cls[(size_t)b * CCH + c] = f2[((size_t)b * (KCT + 1)) * CCH + c] + cls[(size_t)b * CCH + c];
    } else {
        int k = j - 1;
        int ind = inds_i[b * KCT + k];
        for (int c = threadIdx.x; c < CCH; c += 256)
            out_cent[((size_t)b * KCT + k) * CCH + c] =
                f2[((size_t)b * (KCT + 1) + j) * CCH + c] + src[((size_t)b * SLN + ind) * CCH + c];
    }
}

// ---------------------------------------------------------------------------
// generic batched bf16 WMMA GEMM: C[M x N] = A[M x K] * B[K x N]  (f32 accum)
// 256 threads = 8 waves, tile 128x128x32, waves 4(M) x 2(N), wave = 32x64
// Double-buffered LDS; A tile staged with GLOBAL_LOAD_ASYNC_TO_LDS_B128,
// B tile staged via coalesced b128 loads + register transpose.
// ---------------------------------------------------------------------------
__global__ __launch_bounds__(256) void gemm_bf16_kernel(
    const bf16* __restrict__ A, long sA, int lda,
    const bf16* __restrict__ Bm, long sB, int ldb,
    int M, int N, int Kt,
    const float* __restrict__ bias,
    const float* __restrict__ resid, long sR,
    const float* __restrict__ rowDiv, int sDiv,
    float scale, int doGelu,
    float* __restrict__ outF, long sOF,
    bf16* __restrict__ outB, long sOB) {
    __shared__ bf16 As[2][128 * 32];
    __shared__ bf16 Bts[2][128 * 32];
    int bz = blockIdx.z;
    int row0 = blockIdx.y * 128, col0 = blockIdx.x * 128;
    const bf16* Ab = A + (size_t)bz * sA;
    const bf16* Bb = Bm + (size_t)bz * sB;
    int tid = threadIdx.x, wave = tid >> 5, lane = tid & 31;
    int lh = lane >> 4, lm = lane & 15;
    int wm = wave & 3, wn = wave >> 2;
    v8f acc[2][4];
#pragma unroll
    for (int mt = 0; mt < 2; ++mt)
#pragma unroll
        for (int nt = 0; nt < 4; ++nt) acc[mt][nt] = vzero8();

    int arow = tid >> 2, aq = tid & 3;   // A: rows (arow, arow+64), 16B quarter aq
    int kb = tid >> 3, nseg = tid & 7;   // B: k row kb, 16-col segment nseg

    auto stageA = [&](int k0, int buf) {
#pragma unroll
        for (int hh = 0; hh < 2; ++hh) {
            int r = arow + hh * 64;
            int grow = row0 + r;
            if (grow < M) {
                unsigned lds = (unsigned)(uintptr_t)&As[buf][r * 32 + aq * 8];
                async_copy16(lds, Ab + (size_t)grow * lda + k0 + aq * 8);
            } else {
                *(uint4*)&As[buf][r * 32 + aq * 8] = make_uint4(0, 0, 0, 0);
            }
        }
    };
    auto stageB = [&](int k0, int buf) {
        bf16 btmp[16];
        int gn0 = col0 + nseg * 16;
        const bf16* bp = Bb + (size_t)(k0 + kb) * ldb + gn0;
        if (gn0 + 15 < N) {
            *(uint4*)&btmp[0] = *(const uint4*)bp;
            *(uint4*)&btmp[8] = *(const uint4*)(bp + 8);
        } else {
#pragma unroll
            for (int j = 0; j < 16; ++j)
                btmp[j] = (gn0 + j < N) ? bp[j] : (bf16)0.0f;
        }
#pragma unroll
        for (int j = 0; j < 16; ++j)
            Bts[buf][(nseg * 16 + j) * 32 + kb] = btmp[j];
    };

    int nk = Kt / 32;
    stageA(0, 0);
    stageB(0, 0);
    for (int t = 0; t < nk; ++t) {
        int cur = t & 1;
        wait_async0();
        __syncthreads();
        if (t + 1 < nk) {
            stageA((t + 1) * 32, cur ^ 1);
            stageB((t + 1) * 32, cur ^ 1);
            if (t + 2 < nk)
                __builtin_prefetch(Bb + (size_t)((t + 2) * 32 + kb) * ldb + col0 + nseg * 16, 0, 1);
        }
        v16bf af[2], bfr[4];
#pragma unroll
        for (int mt = 0; mt < 2; ++mt)
            af[mt] = lds_afrag(As[cur], wm * 32 + mt * 16, 32, 0, lm, lh);
#pragma unroll
        for (int nt = 0; nt < 4; ++nt)
            bfr[nt] = lds_bfrag(Bts[cur], wn * 64 + nt * 16, 32, 0, lm, lh);
#pragma unroll
        for (int mt = 0; mt < 2; ++mt)
#pragma unroll
            for (int nt = 0; nt < 4; ++nt)
                acc[mt][nt] = wmma_bf16(af[mt], bfr[nt], acc[mt][nt]);
    }

#pragma unroll
    for (int mt = 0; mt < 2; ++mt) {
#pragma unroll
        for (int nt = 0; nt < 4; ++nt) {
            int col = col0 + wn * 64 + nt * 16 + lm;
#pragma unroll
            for (int r = 0; r < 8; ++r) {
                int row = row0 + wm * 32 + mt * 16 + r + lh * 8;
                if (row < M && col < N) {
                    float v = acc[mt][nt][r] * scale;
                    if (bias)   v += bias[col];
                    if (resid)  v += resid[(size_t)bz * sR + (size_t)row * N + col];
                    if (rowDiv) v /= rowDiv[bz * sDiv + row];
                    if (doGelu) v = 0.5f * v * (1.0f + erff(v * 0.70710678118f));
                    if (outF) outF[(size_t)bz * sOF + (size_t)row * N + col] = v;
                    if (outB) outB[(size_t)bz * sOB + (size_t)row * N + col] = (bf16)v;
                }
            }
        }
    }
}

// ---------------------------------------------------------------------------
// flash attention: grid = B*H*16 workgroups of 128 threads (4 waves),
// each wave owns a 16-row q tile; kv processed in 32-column chunks.
// Double-buffered: K chunk staged via async-to-LDS, V chunk via reg transpose.
// ---------------------------------------------------------------------------
__global__ __launch_bounds__(128) void flash_attn_kernel(const bf16* __restrict__ qkv,
                                                         bf16* __restrict__ a_out) {
    int wg = blockIdx.x;
    int qt = wg & 15;
    int h  = (wg >> 4) % NHD;
    int b  = wg / (16 * NHD);
    int tid = threadIdx.x, wave = tid >> 5, lane = tid & 31;
    int lh = lane >> 4, lm = lane & 15;

    __shared__ bf16 Ks[2][32 * 64];      // [kv][d]  == Bt layout for S = q k^T
    __shared__ bf16 VsT[2][64 * 32];     // [d][kv]  == Bt layout for P V
    __shared__ bf16 Pst[4][16 * 32];     // per-wave P staging (C-layout -> A-layout)

    int qbase = qt * 64 + wave * 16;
    const bf16* qrow = qkv + ((size_t)(b * SLN + qbase + lm)) * (3 * CCH) + h * HDD;
    v16bf aq0, aq1;
    *(uint4*)&aq0       = *(const uint4*)(qrow + lh * 8);
    *((uint4*)&aq0 + 1) = *(const uint4*)(qrow + lh * 8 + 16);
    *(uint4*)&aq1       = *(const uint4*)(qrow + 32 + lh * 8);
    *((uint4*)&aq1 + 1) = *(const uint4*)(qrow + 32 + lh * 8 + 16);

    v8f o0 = vzero8(), o1 = vzero8(), o2 = vzero8(), o3 = vzero8();
    float rm[8], rl[8];
#pragma unroll
    for (int r = 0; r < 8; ++r) { rm[r] = -1e30f; rl[r] = 0.0f; }

    int krow = tid >> 3, kseg8 = tid & 7;  // K async: rows (krow, krow+16), 16B seg
    int vrow = tid >> 2, vseg = tid & 3;   // V transpose: row vrow, 16-elem seg

    auto stageKV = [&](int kv, int buf) {
#pragma unroll
        for (int hh = 0; hh < 2; ++hh) {
            int r = krow + hh * 16;
            unsigned lds = (unsigned)(uintptr_t)&Ks[buf][r * 64 + kseg8 * 8];
            async_copy16(lds, qkv + ((size_t)(b * SLN + kv + r)) * (3 * CCH)
                                  + CCH + h * HDD + kseg8 * 8);
        }
        const bf16* vp = qkv + ((size_t)(b * SLN + kv + vrow)) * (3 * CCH)
                             + 2 * CCH + h * HDD + vseg * 16;
        bf16 vloc[16];
        *(uint4*)&vloc[0] = *(const uint4*)vp;
        *(uint4*)&vloc[8] = *(const uint4*)(vp + 8);
#pragma unroll
        for (int j = 0; j < 16; ++j)
            VsT[buf][(vseg * 16 + j) * 32 + vrow] = vloc[j];
    };

    stageKV(0, 0);
    for (int c = 0; c < SLN / 32; ++c) {
        int cur = c & 1;
        wait_async0();
        __syncthreads();
        if (c + 1 < SLN / 32) stageKV((c + 1) * 32, cur ^ 1);

        // S = q @ k^T  (two 16x16 n-tiles)
        v8f s0 = vzero8(), s1 = vzero8();
        s0 = wmma_bf16(aq0, lds_bfrag(Ks[cur], 0, 64, 0, lm, lh), s0);
        s0 = wmma_bf16(aq1, lds_bfrag(Ks[cur], 0, 64, 32, lm, lh), s0);
        s1 = wmma_bf16(aq0, lds_bfrag(Ks[cur], 16, 64, 0, lm, lh), s1);
        s1 = wmma_bf16(aq1, lds_bfrag(Ks[cur], 16, 64, 32, lm, lh), s1);

        // online softmax; stash P through LDS to convert C-layout -> A-layout
#pragma unroll
        for (int r = 0; r < 8; ++r) {
            float v0 = s0[r] * ATT_SCALE;
            float v1 = s1[r] * ATT_SCALE;
            float mx = rowred_max16(fmaxf(v0, v1));
            float nm = fmaxf(rm[r], mx);
            float p0 = __expf(v0 - nm);
            float p1 = __expf(v1 - nm);
            float alpha = __expf(rm[r] - nm);
            rm[r] = nm;
            float ps = rowred_sum16(p0 + p1);
            rl[r] = rl[r] * alpha + ps;
            o0[r] *= alpha; o1[r] *= alpha; o2[r] *= alpha; o3[r] *= alpha;
            Pst[wave][(r + lh * 8) * 32 + lm]      = (bf16)p0;
            Pst[wave][(r + lh * 8) * 32 + lm + 16] = (bf16)p1;
        }
        asm volatile("s_wait_dscnt 0x0" ::: "memory");
        v16bf aP;
        {
            const bf16* pp = &Pst[wave][lm * 32 + lh * 8];
            *(uint4*)&aP       = *(const uint4*)pp;
            *((uint4*)&aP + 1) = *(const uint4*)(pp + 16);
        }
        o0 = wmma_bf16(aP, lds_bfrag(VsT[cur], 0,  32, 0, lm, lh), o0);
        o1 = wmma_bf16(aP, lds_bfrag(VsT[cur], 16, 32, 0, lm, lh), o1);
        o2 = wmma_bf16(aP, lds_bfrag(VsT[cur], 32, 32, 0, lm, lh), o2);
        o3 = wmma_bf16(aP, lds_bfrag(VsT[cur], 48, 32, 0, lm, lh), o3);
    }

#pragma unroll
    for (int r = 0; r < 8; ++r) {
        size_t row = (size_t)(b * SLN + qbase + r + lh * 8);
        float inv = 1.0f / rl[r];
        bf16* op = a_out + row * CCH + h * HDD + lm;
        op[0]  = (bf16)(o0[r] * inv);
        op[16] = (bf16)(o1[r] * inv);
        op[32] = (bf16)(o2[r] * inv);
        op[48] = (bf16)(o3[r] * inv);
    }
}

// ---------------------------------------------------------------------------
// launcher
// ---------------------------------------------------------------------------
extern "C" void kernel_launch(void* const* d_in, const int* in_sizes, int n_in,
                              void* d_out, int out_size, void* d_ws, size_t ws_size,
                              hipStream_t stream) {
    const float* cls      = (const float*)d_in[0];
    const float* src      = (const float*)d_in[1];
    const unsigned char* mask = (const unsigned char*)d_in[2];
    const float* bn_g     = (const float*)d_in[3];
    const float* bn_b     = (const float*)d_in[4];
    const float* qkv_w    = (const float*)d_in[5];
    const float* qkv_b    = (const float*)d_in[6];
    const float* proj_w   = (const float*)d_in[7];
    const float* proj_b   = (const float*)d_in[8];
    const float* blk_bias = (const float*)d_in[9];
    const float* ln1_g    = (const float*)d_in[10];
    const float* ln1_b    = (const float*)d_in[11];
    const float* fc1_w    = (const float*)d_in[12];
    const float* fc1_b    = (const float*)d_in[13];
    const float* ln2_g    = (const float*)d_in[14];
    const float* ln2_b    = (const float*)d_in[15];
    const float* fc2_w    = (const float*)d_in[16];
    const float* fc2_b    = (const float*)d_in[17];

    char* ws = (char*)d_ws;
    size_t off = 0;
    auto alloc = [&](size_t bytes) -> char* {
        char* p = ws + off;
        off += (bytes + 255) & ~(size_t)255;
        return p;
    };

    float* mean_src  = (float*)alloc((size_t)BB * CCH * 4);
    float* samplingT = (float*)alloc((size_t)BB * CCH * NPD * 4);
    int*   inds_i    = (int*)  alloc((size_t)BB * KCT * 4);
    bf16*  h_bf      = (bf16*) alloc((size_t)BB * SLN * CCH * 2);
    bf16*  wq_bf     = (bf16*) alloc((size_t)CCH * 3 * CCH * 2);
    bf16*  qkv_bf    = (bf16*) alloc((size_t)BB * SLN * 3 * CCH * 2);
    bf16*  a_bf      = (bf16*) alloc((size_t)BB * SLN * CCH * 2);
    bf16*  wp_bf     = (bf16*) alloc((size_t)CCH * CCH * 2);
    float* x_f       = (float*)alloc((size_t)BB * SLN * CCH * 4);
    float* xmean     = (float*)alloc((size_t)BB * CCH * 4);
    bf16*  node_bf   = (bf16*) alloc((size_t)BB * SLN * CCH * 2);
    bf16*  centT_bf  = (bf16*) alloc((size_t)BB * CCH * KCT * 2);
    bf16*  assignT   = (bf16*) alloc((size_t)BB * KCT * SLN * 2);
    float* colsum    = (float*)alloc((size_t)BB * KCT * 4);
    bf16*  catln_bf  = (bf16*) alloc((size_t)BB * NPD * CCH * 2);
    bf16*  w1_bf     = (bf16*) alloc((size_t)CCH * 4 * CCH * 2);
    bf16*  f1_bf     = (bf16*) alloc((size_t)BB * NPD * 4 * CCH * 2);
    float* cent_f    = (float*)alloc((size_t)BB * KCT * 4 * CCH * 4);
    bf16*  cat2ln_bf = (bf16*) alloc((size_t)BB * (KCT + 1) * 4 * CCH * 2);
    bf16*  w2_bf     = (bf16*) alloc((size_t)4 * CCH * CCH * 2);
    float* f2_f      = (float*)alloc((size_t)BB * (KCT + 1) * CCH * 4);

    float* out       = (float*)d_out;
    float* out_cls   = out;                                    // B*C
    float* out_cent  = out + (size_t)BB * CCH;                 // B*K*C
    float* out_logit = out_cent + (size_t)BB * KCT * CCH;      // B*SL*K
    float* out_inds  = out_logit + (size_t)BB * SLN * KCT;     // B*K

    // weight conversion f32 -> bf16
    {
        int n;
        n = CCH * 3 * CCH;  f32_to_bf16_kernel<<<(n + 255) / 256, 256, 0, stream>>>(qkv_w, wq_bf, n);
        n = CCH * CCH;      f32_to_bf16_kernel<<<(n + 255) / 256, 256, 0, stream>>>(proj_w, wp_bf, n);
        n = CCH * 4 * CCH;  f32_to_bf16_kernel<<<(n + 255) / 256, 256, 0, stream>>>(fc1_w, w1_bf, n);
        n = 4 * CCH * CCH;  f32_to_bf16_kernel<<<(n + 255) / 256, 256, 0, stream>>>(fc2_w, w2_bf, n);
    }

    // masked mean + sampling points + FPS
    colmean_kernel<<<dim3((CCH + 255) / 256, BB), 256, 0, stream>>>(src, mask, mean_src, SLN, CCH);
    build_samplingT_kernel<<<BB * NPD, 256, 0, stream>>>(src, mask, mean_src, samplingT);
    fps_kernel<<<BB, 1024, 0, stream>>>(samplingT, inds_i, out_inds);

    // attention block
    ln_rows_kernel<<<BB * SLN, 256, 0, stream>>>(src, bn_g, bn_b, h_bf, CCH, 1e-6f);
    gemm_bf16_kernel<<<dim3(18, 64, 1), 256, 0, stream>>>(
        h_bf, 0, CCH, wq_bf, 0, 3 * CCH, BB * SLN, 3 * CCH, CCH,
        qkv_b, nullptr, 0, nullptr, 0, 1.0f, 0, nullptr, 0, qkv_bf, 0);
    flash_attn_kernel<<<BB * NHD * 16, 128, 0, stream>>>(qkv_bf, a_bf);
    gemm_bf16_kernel<<<dim3(6, 64, 1), 256, 0, stream>>>(
        a_bf, 0, CCH, wp_bf, 0, CCH, BB * SLN, CCH, CCH,
        proj_b, src, 0, nullptr, 0, 1.0f, 0, x_f, 0, nullptr, 0);

    // node features -> cosine logits
    colmean_kernel<<<dim3((CCH + 255) / 256, BB), 256, 0, stream>>>(x_f, nullptr, xmean, SLN, CCH);
    node_norm_kernel<<<BB * SLN, 256, 0, stream>>>(x_f, xmean, blk_bias, node_bf);
    gather_centT_kernel<<<BB * KCT, 256, 0, stream>>>(node_bf, inds_i, centT_bf);
    gemm_bf16_kernel<<<dim3(1, 8, BB), 256, 0, stream>>>(
        node_bf, (long)SLN * CCH, CCH, centT_bf, (long)CCH * KCT, KCT,
        SLN, KCT, CCH, nullptr, nullptr, 0, nullptr, 0, 5.0f, 0,
        out_logit, (long)SLN * KCT, nullptr, 0);

    // assignments
    zero_f32_kernel<<<(BB * KCT + 255) / 256, 256, 0, stream>>>(colsum, BB * KCT);
    softmax_assign_kernel<<<BB * SLN / 8, 256, 0, stream>>>(out_logit, assignT, colsum);

    // fc1 (GELU) on concat(cls, src)
    ln_cat_kernel<<<BB * NPD, 256, 0, stream>>>(cls, src, ln1_g, ln1_b, catln_bf, 1e-5f);
    gemm_bf16_kernel<<<dim3(24, 65, 1), 256, 0, stream>>>(
        catln_bf, 0, CCH, w1_bf, 0, 4 * CCH, BB * NPD, 4 * CCH, CCH,
        fc1_b, nullptr, 0, nullptr, 0, 1.0f, 1, nullptr, 0, f1_bf, 0);

    // centroids = assign^T @ f1_src / colsum   (batched, K=SL)
    gemm_bf16_kernel<<<dim3(24, 1, BB), 256, 0, stream>>>(
        assignT, (long)KCT * SLN, SLN,
        f1_bf + 4 * CCH, (long)NPD * 4 * CCH, 4 * CCH,
        KCT, 4 * CCH, SLN,
        nullptr, nullptr, 0, colsum, KCT, 1.0f, 0,
        cent_f, (long)KCT * 4 * CCH, nullptr, 0);

    // fc2 and outputs
    ln_cat2_kernel<<<BB * (KCT + 1), 256, 0, stream>>>(f1_bf, cent_f, ln2_g, ln2_b, cat2ln_bf, 1e-5f);
    gemm_bf16_kernel<<<dim3(6, 5, 1), 256, 0, stream>>>(
        cat2ln_bf, 0, 4 * CCH, w2_bf, 0, CCH, BB * (KCT + 1), CCH, 4 * CCH,
        fc2_b, nullptr, 0, nullptr, 0, 1.0f, 0, f2_f, 0, nullptr, 0);
    final_out_kernel<<<BB * (KCT + 1), 256, 0, stream>>>(f2_f, cls, src, inds_i, out_cls, out_cent);

    (void)in_sizes; (void)n_in; (void)out_size; (void)ws_size;
}
